// MetalMLAAttention_26534307955352
// MI455X (gfx1250) — matrix operand order repaired and checked by reading, not compiled, not run on hardware
//
#include <hip/hip_runtime.h>

// ---------------------------------------------------------------------------
// Problem constants (match the reference)
// ---------------------------------------------------------------------------
#define B_ 2
#define S_ 2048
#define D_ 2048
#define H_ 16
#define NOPE_ 128
#define ROPE_ 64
#define VD_ 128
#define LORA_ 512
#define QH_ (NOPE_ + ROPE_)              // 192
#define QN_ (H_ * QH_)                   // 3072
#define CKVN_ (LORA_ + ROPE_)            // 576
#define KVN_ (H_ * (NOPE_ + VD_))        // 4096
#define ON_ (H_ * VD_)                   // 2048
#define MROWS_ (B_ * S_)                 // 4096

typedef __attribute__((ext_vector_type(16))) __bf16        v16bf;
typedef __attribute__((ext_vector_type(8)))  float         v8f;
typedef __attribute__((ext_vector_type(8)))  unsigned int  v8u;
typedef __attribute__((ext_vector_type(4)))  unsigned int  v4u;
typedef __attribute__((ext_vector_type(8)))  int           v8i_t;
typedef __attribute__((ext_vector_type(4)))  int           v4i_t;

#if __has_builtin(__builtin_amdgcn_tensor_load_to_lds)
#define USE_TDM 1
#else
#define USE_TDM 0
#endif

// ---------------------------------------------------------------------------
// Helpers
// ---------------------------------------------------------------------------
static __device__ __forceinline__ unsigned short f32_to_bf16(float f) {
    unsigned int u = __builtin_bit_cast(unsigned int, f);
    u += 0x7FFFu + ((u >> 16) & 1u);             // round-to-nearest-even
    return (unsigned short)(u >> 16);
}
static __device__ __forceinline__ float bf16_to_f32(unsigned short h) {
    return __builtin_bit_cast(float, ((unsigned int)h) << 16);
}
static __device__ __forceinline__ void store_c(float* p, float v) { *p = v; }
static __device__ __forceinline__ void store_c(unsigned short* p, float v) { *p = f32_to_bf16(v); }

// packed pair loaders (2 consecutive elements -> one u32 of 2 bf16)
static __device__ __forceinline__ unsigned int load_pair_bf16(const float* p) {
    float2 f = *(const float2*)p;
    return (unsigned int)f32_to_bf16(f.x) | ((unsigned int)f32_to_bf16(f.y) << 16);
}
static __device__ __forceinline__ unsigned int load_pair_bf16(const unsigned short* p) {
    return *(const unsigned int*)p;
}

static __device__ __forceinline__ v8f vzero8() {
    v8f v;
#pragma unroll
    for (int i = 0; i < 8; i++) v[i] = 0.0f;
    return v;
}

static __device__ __forceinline__ v8f wmma_bf16(v16bf a, v16bf b, v8f c) {
    // v_wmma_f32_16x16x32_bf16: (neg_a, A, neg_b, B, c_mod, C, reuse_a, reuse_b)
    return __builtin_amdgcn_wmma_f32_16x16x32_bf16(false, a, false, b, (short)0, c, false, false);
}

// A-fragment (16x32 bf16), element (m,k) stored at base[m*stride + k].
// k order used here (B matches): lanes 0-15: M=lane, K=0..15 (VGPR i: K=2i,2i+1)
//                 lanes 16-31: same M, K=16..31.
static __device__ __forceinline__ v16bf load_afrag(const unsigned short* base, int stride) {
    const int lane = threadIdx.x & 31;
    const int m = lane & 15;
    const int hi = (lane >> 4) & 1;
    v8u u;
#pragma unroll
    for (int i = 0; i < 8; i++) {
        int kb = 16 * hi + 2 * i;
        u[i] = *(const unsigned int*)(base + (size_t)m * stride + kb);
    }
    return __builtin_bit_cast(v16bf, u);
}

// B-fragment (32x16 bf16), element (k,n) stored at base[k*stride + n] (n contiguous).
// k order used here (A matches): lanes 0-15: N=lane, K=0..15 (VGPR j: K=2j,2j+1); lanes 16-31: K=16..31.
static __device__ __forceinline__ v16bf load_bfrag(const unsigned short* base, int stride) {
    const int lane = threadIdx.x & 31;
    const int n = lane & 15;
    const int hi = (lane >> 4) & 1;
    v8u u;
#pragma unroll
    for (int j = 0; j < 8; j++) {
        int k = 16 * hi + 2 * j;
        unsigned int lo = base[(size_t)k * stride + n];
        unsigned int hv = base[(size_t)(k + 1) * stride + n];
        u[j] = lo | (hv << 16);
    }
    return __builtin_bit_cast(v16bf, u);
}

// B-fragment where element (k,n) is stored at base[n*stride + k] (k contiguous).
static __device__ __forceinline__ v16bf load_bfrag_t(const unsigned short* base, int stride) {
    const int lane = threadIdx.x & 31;
    const int n = lane & 15;
    const int hi = (lane >> 4) & 1;
    v8u u;
#pragma unroll
    for (int j = 0; j < 8; j++) {
        u[j] = *(const unsigned int*)(base + (size_t)n * stride + 16 * hi + 2 * j);
    }
    return __builtin_bit_cast(v16bf, u);
}

#if USE_TDM
// ---------------------------------------------------------------------------
// Tensor Data Mover: 2-D tile of 16-bit elements, global -> LDS, with LDS row
// padding. pad_interval_code c => pad every 2^(c+1) DWORDs; pad_amount_code a
// => pad a+1 DWORDs. All descriptor inputs here are wave-uniform (SGPRs).
// ---------------------------------------------------------------------------
static __device__ __forceinline__ void tdm_load_2d_b16(
    unsigned int lds_addr_bytes, const void* gaddr,
    unsigned int tile_w, unsigned int tile_h, unsigned int row_stride_elems,
    unsigned int pad_interval_code, unsigned int pad_amount_code)
{
    unsigned long long ga = (unsigned long long)(size_t)gaddr;
    v4u g0;
    g0[0] = 1u;                                              // count=1, user mode
    g0[1] = lds_addr_bytes;                                  // lds_addr
    g0[2] = (unsigned int)(ga & 0xFFFFFFFFu);                // global_addr[31:0]
    g0[3] = ((unsigned int)(ga >> 32) & 0x01FFFFFFu)         // global_addr[56:32]
          | (2u << 30);                                      // type=2 ("image")
    unsigned int w0 = (1u << 16)                             // data_size = 2 bytes
                    | (1u << 20)                             // pad_enable
                    | (pad_interval_code << 22)
                    | (pad_amount_code << 25);
    unsigned int td0 = tile_w;                               // tensor_dim0 == tile width
    unsigned int td1 = tile_h;                               // tensor_dim1 == tile height
    unsigned long long s0 = (unsigned long long)row_stride_elems;
    v8i_t g1;
    g1[0] = (int)w0;
    g1[1] = (int)((td0 & 0xFFFFu) << 16);                    // [47:32]=atomic_addr=0, [63:48]=td0 lo
    g1[2] = (int)(((td0 >> 16) & 0xFFFFu) | ((td1 & 0xFFFFu) << 16));
    g1[3] = (int)(((td1 >> 16) & 0xFFFFu) | ((tile_w & 0xFFFFu) << 16)); // [127:112]=tile_dim0
    g1[4] = (int)(tile_h & 0xFFFFu);                         // [143:128]=tile_dim1; tile_dim2=0
    g1[5] = (int)(unsigned int)(s0 & 0xFFFFFFFFu);           // tensor_dim0_stride lo
    g1[6] = (int)(unsigned int)((s0 >> 32) & 0xFFFFu);       // stride hi; tensor_dim1_stride=0
    g1[7] = 0;
    v4i_t z4; z4[0] = 0; z4[1] = 0; z4[2] = 0; z4[3] = 0;
#if defined(__clang_major__) && (__clang_major__ >= 23)
    v8i_t z8;
#pragma unroll
    for (int i = 0; i < 8; i++) z8[i] = 0;
    __builtin_amdgcn_tensor_load_to_lds(g0, g1, z4, z4, z8, 0);
#else
    __builtin_amdgcn_tensor_load_to_lds(g0, g1, z4, z4, 0);
#endif
}
#endif // USE_TDM

// ---------------------------------------------------------------------------
// Generic GEMM: C[M,N] = A[M,K] @ W[K,N]
// A: f32 or bf16(bits), W: f32 (converted to bf16 on the fly), C: f32 or bf16.
// Block: 256 threads = 8 waves, tile 64M x 128N, K step 32; wave tile 32x32.
// ---------------------------------------------------------------------------
template <typename TA, typename TC>
__global__ __launch_bounds__(256) void gemm_bf16_wmma(
    const TA* __restrict__ A, const float* __restrict__ W, TC* __restrict__ C,
    int M, int N, int K, int lda, int ldc)
{
    const int SA = 34;   // 32 + 2 pad (keeps 32-bit frag loads 4B aligned)
    const int SB = 130;  // 128 + 2 pad
    __shared__ unsigned short As[64 * SA];
    __shared__ unsigned short Bs[32 * SB];

    const int m0 = blockIdx.x * 64;
    const int n0 = blockIdx.y * 128;
    const int tid = threadIdx.x;
    const int wid = tid >> 5;
    const int lane = tid & 31;
    const int wm = (wid >> 2) * 32;   // 0 or 32
    const int wn = (wid & 3) * 32;    // 0,32,64,96

    v8f acc[2][2];
#pragma unroll
    for (int i = 0; i < 2; i++)
#pragma unroll
        for (int j = 0; j < 2; j++) acc[i][j] = vzero8();

    for (int k0 = 0; k0 < K; k0 += 32) {
        // prefetch next K-chunk while this one is staged/consumed
        if (k0 + 32 < K) {
            int pr = tid >> 2;                   // 0..63 rows of A
            __builtin_prefetch(&A[(size_t)(m0 + pr) * lda + k0 + 32 + (tid & 3) * 8], 0, 1);
            int wr = tid >> 3;                   // 0..31 rows of W
            int wc = (tid & 7) * 16;
            if (n0 + wc < N)
                __builtin_prefetch(&W[(size_t)(k0 + 32 + wr) * N + n0 + wc], 0, 1);
        }
        __syncthreads();
        // stage A tile 64x32 as bf16 pairs (4 pairs/thread)
#pragma unroll
        for (int i = 0; i < 4; i++) {
            int e = tid + i * 256;               // pair index, 1024 pairs
            int r = e >> 4, c = (e & 15) * 2;
            *(unsigned int*)&As[r * SA + c] = load_pair_bf16(&A[(size_t)(m0 + r) * lda + k0 + c]);
        }
        // stage B tile 32x128 as bf16 pairs (8 pairs/thread), zero-pad past N
#pragma unroll
        for (int i = 0; i < 8; i++) {
            int e = tid + i * 256;               // pair index, 2048 pairs
            int r = e >> 6, c = (e & 63) * 2;
            unsigned int pk = 0;
            if (n0 + c < N) {                    // N is even: pairs never straddle
                float2 w = *(const float2*)&W[(size_t)(k0 + r) * N + n0 + c];
                pk = (unsigned int)f32_to_bf16(w.x) | ((unsigned int)f32_to_bf16(w.y) << 16);
            }
            *(unsigned int*)&Bs[r * SB + c] = pk;
        }
        __syncthreads();

        v16bf a0 = load_afrag(As + (size_t)wm * SA, SA);
        v16bf a1 = load_afrag(As + (size_t)(wm + 16) * SA, SA);
        v16bf b0 = load_bfrag(Bs + wn, SB);
        v16bf b1 = load_bfrag(Bs + wn + 16, SB);
        acc[0][0] = wmma_bf16(a0, b0, acc[0][0]);
        acc[0][1] = wmma_bf16(a0, b1, acc[0][1]);
        acc[1][0] = wmma_bf16(a1, b0, acc[1][0]);
        acc[1][1] = wmma_bf16(a1, b1, acc[1][1]);
    }

    // Write back: C layout — lane holds column n = lane&15, VGPR r -> row r (+8 for hi half)
    const int hi = (lane >> 4) & 1;
    const int nl = lane & 15;
#pragma unroll
    for (int mi = 0; mi < 2; mi++)
#pragma unroll
        for (int ni = 0; ni < 2; ni++)
#pragma unroll
            for (int r = 0; r < 8; r++) {
                int m = m0 + wm + mi * 16 + r + 8 * hi;
                int n = n0 + wn + ni * 16 + nl;
                if (n < N) store_c(&C[(size_t)m * ldc + n], acc[mi][ni][r]);
            }
}

// ---------------------------------------------------------------------------
// RoPE (in-place on bf16 buffers)
// ---------------------------------------------------------------------------
__global__ void rope_q_kernel(unsigned short* __restrict__ q) {
    int t = blockIdx.x * blockDim.x + threadIdx.x;
    int i = t & 31;  t >>= 5;
    int h = t & (H_ - 1); t >>= 4;
    int row = t;                    // 0..MROWS_-1
    int s = row & (S_ - 1);
    float inv = __powf(10000.0f, -(float)i * (1.0f / 32.0f));
    float ang = (float)s * inv;
    float c = __cosf(ang), sn = __sinf(ang);
    unsigned short* p = q + (size_t)row * QN_ + h * QH_ + NOPE_ + 2 * i;
    float xe = bf16_to_f32(p[0]);
    float xo = bf16_to_f32(p[1]);
    p[0] = f32_to_bf16(xe * c - xo * sn);
    p[1] = f32_to_bf16(xo * c + xe * sn);
}

__global__ void rope_k_kernel(unsigned short* __restrict__ ckv) {
    int t = blockIdx.x * blockDim.x + threadIdx.x;
    int i = t & 31;
    int row = t >> 5;
    int s = row & (S_ - 1);
    float inv = __powf(10000.0f, -(float)i * (1.0f / 32.0f));
    float ang = (float)s * inv;
    float c = __cosf(ang), sn = __sinf(ang);
    unsigned short* p = ckv + (size_t)row * CKVN_ + LORA_ + 2 * i;
    float xe = bf16_to_f32(p[0]);
    float xo = bf16_to_f32(p[1]);
    p[0] = f32_to_bf16(xe * c - xo * sn);
    p[1] = f32_to_bf16(xo * c + xe * sn);
}

// ---------------------------------------------------------------------------
// Causal flash attention.
// Grid: (S/64, H, B); block = 128 threads = 4 waves; each wave owns 16 q rows.
// q: bf16 [MROWS_][3072], kv: bf16 [MROWS_][4096] (per head: 128 k_nope ++ 128 v),
// ckv: bf16 [MROWS_][576] (k_pe at 512..575, roped), out: bf16 [MROWS_][2048].
// K/V tiles staged via Tensor Data Mover (TENSORcnt) when available.
// ---------------------------------------------------------------------------
__global__ __launch_bounds__(128) void mla_attn_kernel(
    const unsigned short* __restrict__ qb,
    const unsigned short* __restrict__ kvb,
    const unsigned short* __restrict__ ckv,
    unsigned short* __restrict__ ob)
{
    const int SK = 194;  // 192 + 2 pad
    const int SV = 130;  // 128 + 2 pad
    __shared__ unsigned short Kt[64 * SK];      // [k row][192 dims]
    __shared__ unsigned short Vt[64 * SV];      // [k row][128 vd]
    __shared__ unsigned short Pt[4][16 * 64];   // per-wave P tile [16 q][64 k]

    const int qt = blockIdx.x;
    const int h  = blockIdx.y;
    const int b  = blockIdx.z;
    const int tid = threadIdx.x;
    const int wid = tid >> 5;
    const int lane = tid & 31;
    const int hi = (lane >> 4) & 1;
    const int nl = lane & 15;

    const int q0 = qt * 64;
    const int qw = q0 + wid * 16;                 // wave's first q row (in-sequence)
    const size_t qrow0 = (size_t)b * S_ + qw;     // global row index

    // Preload Q fragments (6 chunks of 32 over the 192-dim head)
    v16bf qf[6];
    const unsigned short* qbase = qb + qrow0 * QN_ + h * QH_;
#pragma unroll
    for (int c = 0; c < 6; c++) qf[c] = load_afrag(qbase + c * 32, QN_);

    v8f oacc[8];
#pragma unroll
    for (int vt = 0; vt < 8; vt++) oacc[vt] = vzero8();
    float mrow[8], lrow[8];
#pragma unroll
    for (int r = 0; r < 8; r++) { mrow[r] = -3.0e38f; lrow[r] = 0.0f; }

    const float scale = rsqrtf((float)QH_);       // 1/sqrt(192)

    for (int kt = 0; kt <= qt; kt++) {
        __syncthreads();
        const size_t krow0 = (size_t)b * S_ + kt * 64;
#if USE_TDM
        if (wid == 0) {
            const unsigned int kt_base = (unsigned int)(size_t)&Kt[0];
            const unsigned int vt_base = (unsigned int)(size_t)&Vt[0];
            // k_nope: 64 x 128, LDS rows padded 64DW interval (c=5) + 33DW (a=32) -> stride 194
            tdm_load_2d_b16(kt_base, kvb + krow0 * KVN_ + h * (NOPE_ + VD_),
                            128, 64, KVN_, 5, 32);
            // k_pe: 64 x 64 into Kt cols 128..191, 32DW interval (c=4) + 65DW (a=64) -> stride 194
            tdm_load_2d_b16(kt_base + NOPE_ * 2, ckv + krow0 * CKVN_ + LORA_,
                            64, 64, CKVN_, 4, 64);
            // V: 64 x 128, 64DW interval (c=5) + 1DW (a=0) -> stride 130
            tdm_load_2d_b16(vt_base, kvb + krow0 * KVN_ + h * (NOPE_ + VD_) + NOPE_,
                            128, 64, KVN_, 5, 0);
            __builtin_amdgcn_s_wait_tensorcnt(0);
        }
#else
        for (int e = tid; e < 64 * 192; e += 128) {
            int r = e / 192, d = e - r * 192;
            unsigned short v;
            if (d < NOPE_) v = kvb[(krow0 + r) * KVN_ + h * (NOPE_ + VD_) + d];
            else           v = ckv[(krow0 + r) * CKVN_ + LORA_ + (d - NOPE_)];
            Kt[r * SK + d] = v;
        }
        for (int e = tid; e < 64 * 128; e += 128) {
            int r = e >> 7, d = e & 127;
            Vt[r * SV + d] = kvb[(krow0 + r) * KVN_ + h * (NOPE_ + VD_) + NOPE_ + d];
        }
#endif
        __syncthreads();

        // Scores: S = Q (16x192) @ K^T (192x64); Kt is [k][d] so (d,n) lives at Kt[n*SK+d]
        v8f sacc[4];
#pragma unroll
        for (int nt = 0; nt < 4; nt++) sacc[nt] = vzero8();
#pragma unroll
        for (int nt = 0; nt < 4; nt++) {
#pragma unroll
            for (int c = 0; c < 6; c++) {
                v16bf bk = load_bfrag_t(Kt + (size_t)(nt * 16) * SK + c * 32, SK);
                sacc[nt] = wmma_bf16(qf[c], bk, sacc[nt]);
            }
        }

        // Online softmax. Lane holds column k = kt*64 + nt*16 + nl,
        // rows q = qw + r + 8*hi (each 16-lane half holds the same 8 rows).
#pragma unroll
        for (int r = 0; r < 8; r++) {
            int qi = qw + r + 8 * hi;
            float sv[4];
            float vmax = -3.0e38f;
#pragma unroll
            for (int nt = 0; nt < 4; nt++) {
                int ki = kt * 64 + nt * 16 + nl;
                float s = sacc[nt][r] * scale;
                s = (ki <= qi) ? s : -3.0e38f;
                sv[nt] = s;
                vmax = fmaxf(vmax, s);
            }
#pragma unroll
            for (int m = 1; m < 16; m <<= 1) vmax = fmaxf(vmax, __shfl_xor(vmax, m, 16));
            float mnew = fmaxf(mrow[r], vmax);
            float corr = __expf(mrow[r] - mnew);
            mrow[r] = mnew;
            float psum = 0.0f;
#pragma unroll
            for (int nt = 0; nt < 4; nt++) {
                float p = __expf(sv[nt] - mnew);
                psum += p;
                Pt[wid][(r + 8 * hi) * 64 + nt * 16 + nl] = f32_to_bf16(p);
            }
#pragma unroll
            for (int m = 1; m < 16; m <<= 1) psum += __shfl_xor(psum, m, 16);
            lrow[r] = lrow[r] * corr + psum;
#pragma unroll
            for (int vt = 0; vt < 8; vt++) oacc[vt][r] *= corr;
        }

        // O += P (16x64) @ V (64x128). P written+read by the same wave; per-wave
        // LDS ops are in-order, no barrier needed.
#pragma unroll
        for (int kc = 0; kc < 2; kc++) {
            v16bf pa = load_afrag(&Pt[wid][kc * 32], 64);
#pragma unroll
            for (int vt = 0; vt < 8; vt++) {
                v16bf bv = load_bfrag(Vt + (size_t)(kc * 32) * SV + vt * 16, SV);
                oacc[vt] = wmma_bf16(pa, bv, oacc[vt]);
            }
        }
    }

    // Normalize and write O (bf16) at [row][h*128 + vd]
#pragma unroll
    for (int r = 0; r < 8; r++) {
        float inv = 1.0f / lrow[r];
        size_t row = qrow0 + r + 8 * hi;
#pragma unroll
        for (int vt = 0; vt < 8; vt++) {
            ob[row * ON_ + h * VD_ + vt * 16 + nl] = f32_to_bf16(oacc[vt][r] * inv);
        }
    }
}

// ---------------------------------------------------------------------------
// Launch
// ---------------------------------------------------------------------------
extern "C" void kernel_launch(void* const* d_in, const int* in_sizes, int n_in,
                              void* d_out, int out_size, void* d_ws, size_t ws_size,
                              hipStream_t stream) {
    (void)in_sizes; (void)n_in; (void)out_size; (void)ws_size;
    const float* x     = (const float*)d_in[0];
    const float* wq    = (const float*)d_in[1];
    const float* wkv_a = (const float*)d_in[2];
    const float* wkv_b = (const float*)d_in[3];
    const float* wo    = (const float*)d_in[4];
    float* out = (float*)d_out;

    unsigned short* qbuf   = (unsigned short*)d_ws;                 // [4096][3072]
    unsigned short* ckvbuf = qbuf   + (size_t)MROWS_ * QN_;         // [4096][576]
    unsigned short* kvbuf  = ckvbuf + (size_t)MROWS_ * CKVN_;       // [4096][4096]
    unsigned short* abuf   = kvbuf  + (size_t)MROWS_ * KVN_;        // [4096][2048]

    dim3 blk(256);
    // q = x @ wq   (M=4096, N=3072, K=2048)
    gemm_bf16_wmma<float, unsigned short>
        <<<dim3(MROWS_ / 64, QN_ / 128), blk, 0, stream>>>(x, wq, qbuf, MROWS_, QN_, D_, D_, QN_);
    // ckv = x @ wkv_a   (N=576, tail-guarded)
    gemm_bf16_wmma<float, unsigned short>
        <<<dim3(MROWS_ / 64, (CKVN_ + 127) / 128), blk, 0, stream>>>(x, wkv_a, ckvbuf, MROWS_, CKVN_, D_, D_, CKVN_);
    // RoPE on q_pe and k_pe
    rope_q_kernel<<<(MROWS_ * H_ * 32) / 256, 256, 0, stream>>>(qbuf);
    rope_k_kernel<<<(MROWS_ * 32) / 256, 256, 0, stream>>>(ckvbuf);
    // kv = c @ wkv_b   (K=512, A has row stride 576)
    gemm_bf16_wmma<unsigned short, unsigned short>
        <<<dim3(MROWS_ / 64, KVN_ / 128), blk, 0, stream>>>(ckvbuf, wkv_b, kvbuf, MROWS_, KVN_, LORA_, CKVN_, KVN_);
    // attention
    mla_attn_kernel<<<dim3(S_ / 64, H_, B_), dim3(128), 0, stream>>>(qbuf, kvbuf, ckvbuf, abuf);
    // out = attn @ wo
    gemm_bf16_wmma<unsigned short, float>
        <<<dim3(MROWS_ / 64, D_ / 128), blk, 0, stream>>>(abuf, wo, out, MROWS_, D_, ON_, ON_, D_);
}
